// FlowNetCEncoder_90151363543757
// MI455X (gfx1250) — matrix operand order repaired and checked by reading, not compiled, not run on hardware
//
#include <hip/hip_runtime.h>

typedef _Float16 h16;
typedef __attribute__((ext_vector_type(16))) _Float16 v16h;
typedef __attribute__((ext_vector_type(8)))  float    v8f;

#define BN_EPS 1e-5f
#define BCOLS   64
#define BSTRIDE 40   // halves per staged column: 32 data + 8 pad (bank-conflict-free)

union AFrag { uint4 u[2]; v16h v; };
union H8    { uint4 u; h16 h[8]; };

// ---------------- weight / input packing ----------------

// OIHW fp32 -> [Cout][Kp] f16, K order = (r*kw+s)*Cin + ci (matches NHWC im2col)
__global__ void pack_conv_weights(const float* __restrict__ w, h16* __restrict__ wt,
                                  int Cout, int Cin, int kh, int kw, int Kp) {
  long idx = (long)blockIdx.x * blockDim.x + threadIdx.x;
  long total = (long)Cout * Kp;
  if (idx >= total) return;
  int k  = (int)(idx % Kp);
  int co = (int)(idx / Kp);
  int K = Cin * kh * kw;
  float val = 0.f;
  if (k < K) {
    int ci   = k % Cin;
    int cell = k / Cin;
    int s = cell % kw;
    int r = cell / kw;
    val = w[(((long)co * Cin + ci) * kh + r) * kw + s];
  }
  wt[(long)co * Kp + k] = (h16)val;
}

// head_w (K, Cout) fp32 -> [Cout][K] f16 (transpose)
__global__ void pack_head_weights(const float* __restrict__ w, h16* __restrict__ wt,
                                  int K, int Cout) {
  long idx = (long)blockIdx.x * blockDim.x + threadIdx.x;
  long total = (long)Cout * K;
  if (idx >= total) return;
  int k  = (int)(idx % K);
  int co = (int)(idx / K);
  wt[(long)co * K + k] = (h16)w[(long)k * Cout + co];
}

// x (N,6,H,W) fp32 NCHW -> img1/img2 (N,H,W,3) f16 NHWC
__global__ void split_input(const float* __restrict__ x, h16* __restrict__ i1,
                            h16* __restrict__ i2, int N, int H, int W) {
  long idx = (long)blockIdx.x * blockDim.x + threadIdx.x;
  long total = (long)N * H * W * 3;
  if (idx >= total) return;
  int c = (int)(idx % 3); long r = idx / 3;
  int w = (int)(r % W); r /= W;
  int h = (int)(r % H); long n = r / H;
  long planes = (long)H * W;
  long hw = (long)h * W + w;
  i1[idx] = (h16)x[(n * 6 + c) * planes + hw];
  i2[idx] = (h16)x[(n * 6 + 3 + c) * planes + hw];
}

// ---------------- WMMA implicit-GEMM convolution ----------------
// Workgroup (128 thr = 4 waves) computes 64(Cout) x 64(spatial).
// B (im2col K-slice, 32x64 f16) double-buffered in LDS, shared by all 4 waves.
// Software pipeline: async DMA for step i+1 overlaps WMMA of step i:
//   loop: s_wait_asynccnt 0 -> barrier -> stage(buf^1) -> compute(buf)
// fast path (Cin%32==0): per-lane global_load_async_to_lds_b128, incremental im2col walk
// slow path: register gather + ds_store_b128.
// mode 0: BN-fold + LeakyReLU, f16 NHWC out.  mode 1: +bias, f32 out.
__global__ __launch_bounds__(128) void conv_wmma(
    const h16* __restrict__ in, const h16* __restrict__ wt,
    h16* __restrict__ out, float* __restrict__ outf,
    const float* __restrict__ gg, const float* __restrict__ bb,
    const float* __restrict__ mm, const float* __restrict__ vv,
    const float* __restrict__ bias,
    int N, int Hin, int Win, int Cin,
    int Cout, int Hout, int Wout,
    int kh, int kw, int stride, int pad, int K, int Kp, int mode, int fast)
{
  __shared__ h16 Bs[2][BCOLS * BSTRIDE];

  const int tid    = threadIdx.x;
  const int lane   = tid & 31;
  const int laneLo = lane & 15;
  const int laneHi = lane >> 4;
  const int wm     = tid >> 5;                    // wave id = M sub-tile
  const int nTilesM = (Cout + 63) >> 6;
  const long Nspat  = (long)N * Hout * Wout;
  const long tileN  = blockIdx.x / nTilesM;
  const int  tileM  = (int)(blockIdx.x % nTilesM);
  const int  co_base = tileM * 64 + wm * 16;
  const bool doCompute = (co_base + 16) <= Cout;  // wave-uniform

  // -------- staging role: this thread owns column jcol, K-half kh2 --------
  const int jcol = tid & 63;
  const int kh2  = tid >> 6;                      // 0/1 -> K 0..15 / 16..31
  long sj   = tileN * 64 + jcol;
  bool jval = (sj < Nspat);
  long ssj  = jval ? sj : 0;
  int  jow  = (int)(ssj % Wout);
  long jr2  = ssj / Wout;
  int  joh  = (int)(jr2 % Hout);
  long jnb  = jr2 / Hout;
  int  jh0  = joh * stride - pad;
  int  jw0  = jow * stride - pad;
  const long jrow = jnb * Hin;                    // (n*Hin), reused each step

  // incremental im2col walk state for the fast path (Cin % 32 == 0 => Kp == K,
  // each +32 K either stays in the cell or wraps to exactly the next cell)
  int st_ci = kh2 * 16, st_sx = 0, st_ry = 0;

  auto stageFast = [&](int buf) {
    int ih = jh0 + st_ry, iw = jw0 + st_sx;
    bool ok = jval & (ih >= 0) & (ih < Hin) & (iw >= 0) & (iw < Win);
    if (ok) {
      const h16* src = in + ((jrow + ih) * (long)Win + iw) * Cin + st_ci;
      unsigned ldsoff = (unsigned)((buf * (BCOLS * BSTRIDE) + jcol * BSTRIDE + kh2 * 16) * sizeof(h16));
      unsigned long ga = (unsigned long)src;
      asm volatile("global_load_async_to_lds_b128 %0, %1, off\n\t"
                   "global_load_async_to_lds_b128 %0, %1, off offset:16"
                   :: "v"(ldsoff), "v"(ga) : "memory");
    } else {
      h16* bcol = &Bs[buf][jcol * BSTRIDE + kh2 * 16];
      uint4 z = {0u, 0u, 0u, 0u};
      *(uint4*)bcol       = z;
      *((uint4*)bcol + 1) = z;
    }
    st_ci += 32;
    if (st_ci >= Cin) { st_ci -= Cin; if (++st_sx == kw) { st_sx = 0; ++st_ry; } }
  };

  auto stageSlow = [&](int buf, int kk) {
    int ci = kk % Cin, cell = kk / Cin;
    int sx = cell % kw, ry = cell / kw;
    H8 lo, hi;
#pragma unroll
    for (int j = 0; j < 16; ++j) {
      h16 val = (h16)0.f;
      int ih = jh0 + ry, iw = jw0 + sx;
      if (((kk + j) < K) & jval & (ih >= 0) & (ih < Hin) & (iw >= 0) & (iw < Win))
        val = in[((jrow + ih) * (long)Win + iw) * Cin + ci];
      if (j < 8) lo.h[j] = val; else hi.h[j - 8] = val;
      if (++ci == Cin) { ci = 0; if (++sx == kw) { sx = 0; ++ry; } }
    }
    h16* bcol = &Bs[buf][jcol * BSTRIDE + kh2 * 16];
    *(uint4*)bcol       = lo.u;
    *((uint4*)bcol + 1) = hi.u;
  };

  // -------- compute role: columns for fragments / stores --------
  long sidx[4]; bool sval[4];
#pragma unroll
  for (int t = 0; t < 4; ++t) {
    sidx[t] = tileN * 64 + t * 16 + laneLo;
    sval[t] = (sidx[t] < Nspat);
  }

  const h16* __restrict__ wrow = wt + (long)(co_base + laneLo) * Kp;
  const int aoff = laneHi ? 8 : 0;

  v8f c0 = {}, c1 = {}, c2 = {}, c3 = {};

  // prologue: stage K-step 0 into buffer 0
  if (fast) stageFast(0); else stageSlow(0, kh2 * 16);

  int it = 0;
  for (int k0 = 0; k0 < Kp; k0 += 32, ++it) {
    const int buf = it & 1;
    if (fast) asm volatile("s_wait_asynccnt 0x0" ::: "memory");  // retire step-i copies
    __syncthreads();

    // issue DMA for the NEXT K-step; overlaps with this step's WMMAs
    if (k0 + 32 < Kp) {
      if (fast) stageFast(buf ^ 1);
      else      stageSlow(buf ^ 1, k0 + 32 + kh2 * 16);
    }

    if (doCompute) {
      // A fragment: lanes 0-15 hold K {0..15}, lanes 16-31 hold K {16..31}
      AFrag a;
      a.u[0] = *(const uint4*)(wrow + k0 + laneHi * 16);
      a.u[1] = *(const uint4*)(wrow + k0 + laneHi * 16 + 8);
      if (k0 + 32 < Kp) __builtin_prefetch(wrow + k0 + 32, 0, 1);  // global_prefetch_b8

#pragma unroll
      for (int t = 0; t < 4; ++t) {
        const h16* bp = &Bs[buf][(t * 16 + laneLo) * BSTRIDE + laneHi * 16];
        AFrag bf;
        bf.u[0] = *(const uint4*)bp;        // ds_load_b128, conflict-free (stride 80B)
        bf.u[1] = *(const uint4*)(bp + 8);
        v8f acc = (t == 0) ? c0 : (t == 1) ? c1 : (t == 2) ? c2 : c3;
        acc = __builtin_amdgcn_wmma_f32_16x16x32_f16(false, a.v, false, bf.v,
                                                     (short)0, acc, false, false);
        if (t == 0) c0 = acc; else if (t == 1) c1 = acc; else if (t == 2) c2 = acc; else c3 = acc;
      }
    }
  }

  if (!doCompute) return;

  // epilogue: folded BN scale/shift (mode 0) or bias (mode 1)
  float sc[8], sh[8];
#pragma unroll
  for (int i = 0; i < 8; ++i) {
    int co = co_base + aoff + i;
    if (mode == 0) {
      float inv = rsqrtf(vv[co] + BN_EPS);
      sc[i] = gg[co] * inv;
      sh[i] = bb[co] - mm[co] * sc[i];
    } else {
      sc[i] = 1.f;
      sh[i] = bias[co];
    }
  }

#pragma unroll
  for (int t = 0; t < 4; ++t) {
    if (!sval[t]) continue;
    v8f acc = (t == 0) ? c0 : (t == 1) ? c1 : (t == 2) ? c2 : c3;
    if (mode == 0) {
      H8 o;
#pragma unroll
      for (int i = 0; i < 8; ++i) {
        float y = acc[i] * sc[i] + sh[i];
        y = (y > 0.f) ? y : 0.1f * y;      // LeakyReLU(0.1)
        o.h[i] = (h16)y;
      }
      *(uint4*)(out + sidx[t] * Cout + co_base + aoff) = o.u;  // packed b128 store
    } else {
#pragma unroll
      for (int i = 0; i < 8; ++i)
        outf[sidx[t] * Cout + co_base + aoff + i] = acc[i] + sh[i];
    }
  }
}

// ---------------- correlation (441 displacements, fused 2x avgpool, /C) ----------------
__global__ __launch_bounds__(256) void corr_kernel(
    const h16* __restrict__ f1, const h16* __restrict__ f2, h16* __restrict__ cat,
    int N, int H, int W, int C, int OH, int OW, int CatC)
{
  long idx = (long)blockIdx.x * blockDim.x + threadIdx.x;
  long total = (long)N * 441 * OH * OW;
  if (idx >= total) return;
  int ow = (int)(idx % OW); long r = idx / OW;
  int oh = (int)(r % OH); r /= OH;
  int d  = (int)(r % 441); long n = r / 441;
  int dx = d / 21 - 10;
  int dy = d % 21 - 10;
  float acc = 0.f;
#pragma unroll
  for (int i = 0; i < 2; ++i)
#pragma unroll
    for (int j = 0; j < 2; ++j) {
      int h = 2 * oh + i, w = 2 * ow + j;
      int h2 = h + dy, w2 = w + dx;
      if (h2 < 0 || h2 >= H || w2 < 0 || w2 >= W) continue;
      const h16* p1 = f1 + ((n * H + h) * (long)W + w) * C;
      const h16* p2 = f2 + ((n * H + h2) * (long)W + w2) * C;
#pragma unroll 8
      for (int c = 0; c < C; ++c) acc += (float)p1[c] * (float)p2[c];
    }
  cat[((n * OH + oh) * (long)OW + ow) * CatC + d] = (h16)(acc * (0.25f / (float)C));
}

// ---------------- 2x2 average pool into the concat buffer ----------------
__global__ void pool_concat(const h16* __restrict__ rin, h16* __restrict__ cat,
                            int N, int H, int W, int C, int CatC, int off)
{
  long idx = (long)blockIdx.x * blockDim.x + threadIdx.x;
  int OH = H >> 1, OW = W >> 1;
  long total = (long)N * OH * OW * C;
  if (idx >= total) return;
  int c  = (int)(idx % C); long r = idx / C;
  int ow = (int)(r % OW); r /= OW;
  int oh = (int)(r % OH); long n = r / OH;
  float s = 0.f;
#pragma unroll
  for (int i = 0; i < 2; ++i)
#pragma unroll
    for (int j = 0; j < 2; ++j)
      s += (float)rin[((n * H + 2 * oh + i) * (long)W + 2 * ow + j) * C + c];
  cat[((n * OH + oh) * (long)OW + ow) * CatC + off + c] = (h16)(s * 0.25f);
}

// conv6 out NHWC (n,4,2,1024) -> NCHW-flat (n, 8192) to match reference reshape
__global__ void repack_head(const h16* __restrict__ b6, h16* __restrict__ hin, int N) {
  long idx = (long)blockIdx.x * blockDim.x + threadIdx.x;
  long total = (long)N * 8192;
  if (idx >= total) return;
  int k = (int)(idx % 8192); long n = idx / 8192;
  int c = k >> 3; int rem = k & 7; int y = rem >> 1; int xw = rem & 1;
  hin[idx] = b6[(((n * 4) + y) * 2 + xw) * (long)1024 + c];
}

// ---------------- host orchestration ----------------
extern "C" void kernel_launch(void* const* d_in, const int* in_sizes, int n_in,
                              void* d_out, int out_size, void* d_ws, size_t ws_size,
                              hipStream_t stream) {
  (void)in_sizes; (void)n_in; (void)out_size; (void)ws_size;
  const float* x = (const float*)d_in[0];
  // params order: conv1,conv2,conv3,redir,conv3_1,conv4,conv4_1,conv5,conv5_1,conv6 x {w,g,b,m,v}
  auto pw = [&](int L){ return (const float*)d_in[1 + L * 5 + 0]; };
  auto pg = [&](int L){ return (const float*)d_in[1 + L * 5 + 1]; };
  auto pb = [&](int L){ return (const float*)d_in[1 + L * 5 + 2]; };
  auto pm = [&](int L){ return (const float*)d_in[1 + L * 5 + 3]; };
  auto pv = [&](int L){ return (const float*)d_in[1 + L * 5 + 4]; };
  const float* head_w = (const float*)d_in[51];
  const float* head_b = (const float*)d_in[52];

  auto kpad = [](int K) { return (K + 31) & ~31; };

  char* base = (char*)d_ws;
  size_t off = 0;
  auto alloc = [&](size_t elems) -> h16* {
    h16* p = (h16*)(base + off);
    off = (off + elems * sizeof(h16) + 255) & ~(size_t)255;
    return p;
  };

  // packed weights (f16)
  h16* wt1  = alloc((size_t)64   * kpad(3 * 49));
  h16* wt2  = alloc((size_t)128  * kpad(64 * 25));
  h16* wt3  = alloc((size_t)256  * kpad(128 * 25));
  h16* wtr  = alloc((size_t)32   * kpad(256));
  h16* wt31 = alloc((size_t)256  * kpad(473 * 9));
  h16* wt4  = alloc((size_t)512  * kpad(256 * 9));
  h16* wt41 = alloc((size_t)512  * kpad(512 * 9));
  h16* wt5  = alloc((size_t)512  * kpad(512 * 9));
  h16* wt51 = alloc((size_t)512  * kpad(512 * 9));
  h16* wt6  = alloc((size_t)1024 * kpad(512 * 9));
  h16* wth  = alloc((size_t)512  * 8192);
  // activations (f16 NHWC)
  h16* img1 = alloc((size_t)32 * 512 * 256 * 3);
  h16* img2 = alloc((size_t)32 * 512 * 256 * 3);
  h16* a1   = alloc((size_t)32 * 256 * 128 * 64);
  h16* a2   = alloc((size_t)32 * 128 * 64 * 128);
  h16* f1   = alloc((size_t)32 * 64 * 32 * 256);
  h16* f2   = alloc((size_t)32 * 64 * 32 * 256);
  h16* rtmp = alloc((size_t)32 * 64 * 32 * 32);
  h16* cat  = alloc((size_t)32 * 32 * 16 * 473);
  h16* b31  = alloc((size_t)32 * 32 * 16 * 256);
  h16* b4   = alloc((size_t)32 * 16 * 8 * 512);
  h16* b41  = alloc((size_t)32 * 16 * 8 * 512);
  h16* b5   = alloc((size_t)32 * 8 * 4 * 512);
  h16* b51  = alloc((size_t)32 * 8 * 4 * 512);
  h16* b6   = alloc((size_t)32 * 4 * 2 * 1024);
  h16* hin  = alloc((size_t)32 * 8192);

  auto pack = [&](const float* w, h16* dst, int Cout, int Cin, int kh, int kw) {
    int Kp = kpad(Cin * kh * kw);
    long total = (long)Cout * Kp;
    pack_conv_weights<<<(int)((total + 255) / 256), 256, 0, stream>>>(w, dst, Cout, Cin, kh, kw, Kp);
  };
  pack(pw(0), wt1,  64,   3,   7, 7);
  pack(pw(1), wt2,  128,  64,  5, 5);
  pack(pw(2), wt3,  256,  128, 5, 5);
  pack(pw(3), wtr,  32,   256, 1, 1);
  pack(pw(4), wt31, 256,  473, 3, 3);
  pack(pw(5), wt4,  512,  256, 3, 3);
  pack(pw(6), wt41, 512,  512, 3, 3);
  pack(pw(7), wt5,  512,  512, 3, 3);
  pack(pw(8), wt51, 512,  512, 3, 3);
  pack(pw(9), wt6,  1024, 512, 3, 3);
  {
    long total = (long)512 * 8192;
    pack_head_weights<<<(int)((total + 255) / 256), 256, 0, stream>>>(head_w, wth, 8192, 512);
  }
  {
    long total = (long)32 * 512 * 256 * 3;
    split_input<<<(int)((total + 255) / 256), 256, 0, stream>>>(x, img1, img2, 32, 512, 256);
  }

  auto conv = [&](const h16* in, const h16* wtp, h16* out, float* outf,
                  int layer, const float* bias,
                  int N, int Hin, int Win, int Cin, int Cout, int Hout, int Wout,
                  int kh, int kw, int stride, int pad, int mode) {
    int K = Cin * kh * kw, Kp = kpad(K);
    long Nspat = (long)N * Hout * Wout;
    int blocks = (int)(((long)((Cout + 63) / 64)) * ((Nspat + 63) / 64));
    int fast = (Cin % 32 == 0) ? 1 : 0;
    const float *g = nullptr, *b = nullptr, *m = nullptr, *v = nullptr;
    if (mode == 0) { g = pg(layer); b = pb(layer); m = pm(layer); v = pv(layer); }
    conv_wmma<<<blocks, 128, 0, stream>>>(in, wtp, out, outf, g, b, m, v, bias,
        N, Hin, Win, Cin, Cout, Hout, Wout, kh, kw, stride, pad, K, Kp, mode, fast);
  };

  // shared feature trunk, image 1 then image 2 (reuse a1/a2 scratch)
  conv(img1, wt1, a1, nullptr, 0, nullptr, 32, 512, 256, 3,   64,  256, 128, 7, 7, 2, 3, 0);
  conv(a1,   wt2, a2, nullptr, 1, nullptr, 32, 256, 128, 64,  128, 128, 64,  5, 5, 2, 2, 0);
  conv(a2,   wt3, f1, nullptr, 2, nullptr, 32, 128, 64,  128, 256, 64,  32,  5, 5, 2, 2, 0);
  conv(img2, wt1, a1, nullptr, 0, nullptr, 32, 512, 256, 3,   64,  256, 128, 7, 7, 2, 3, 0);
  conv(a1,   wt2, a2, nullptr, 1, nullptr, 32, 256, 128, 64,  128, 128, 64,  5, 5, 2, 2, 0);
  conv(a2,   wt3, f2, nullptr, 2, nullptr, 32, 128, 64,  128, 256, 64,  32,  5, 5, 2, 2, 0);

  // redirection 1x1 conv, correlation, avgpool -> concat (473 ch NHWC)
  conv(f1, wtr, rtmp, nullptr, 3, nullptr, 32, 64, 32, 256, 32, 64, 32, 1, 1, 1, 0, 0);
  {
    long total = (long)32 * 441 * 32 * 16;
    corr_kernel<<<(int)((total + 255) / 256), 256, 0, stream>>>(f1, f2, cat, 32, 64, 32, 256, 32, 16, 473);
  }
  {
    long total = (long)32 * 32 * 16 * 32;
    pool_concat<<<(int)((total + 255) / 256), 256, 0, stream>>>(rtmp, cat, 32, 64, 32, 32, 473, 441);
  }

  // deep trunk
  conv(cat, wt31, b31, nullptr, 4, nullptr, 32, 32, 16, 473, 256,  32, 16, 3, 3, 1, 1, 0);
  conv(b31, wt4,  b4,  nullptr, 5, nullptr, 32, 32, 16, 256, 512,  16, 8,  3, 3, 2, 1, 0);
  conv(b4,  wt41, b41, nullptr, 6, nullptr, 32, 16, 8,  512, 512,  16, 8,  3, 3, 1, 1, 0);
  conv(b41, wt5,  b5,  nullptr, 7, nullptr, 32, 16, 8,  512, 512,  8,  4,  3, 3, 2, 1, 0);
  conv(b5,  wt51, b51, nullptr, 8, nullptr, 32, 8,  4,  512, 512,  8,  4,  3, 3, 1, 1, 0);
  conv(b51, wt6,  b6,  nullptr, 9, nullptr, 32, 8,  4,  512, 1024, 4,  2,  3, 3, 2, 1, 0);

  // head GEMM: (32, 8192) @ (8192, 512) + b  -> f32 d_out
  {
    long total = (long)32 * 8192;
    repack_head<<<(int)((total + 255) / 256), 256, 0, stream>>>(b6, hin, 32);
  }
  conv(hin, wth, nullptr, (float*)d_out, -1, head_b, 32, 1, 1, 8192, 512, 1, 1, 1, 1, 1, 0, 1);
}
